// MHA_50955492000029
// MI455X (gfx1250) — compile-verified, matrix operand order checked by reading, not run on hardware
//
#include <hip/hip_runtime.h>
#include <hip/hip_bf16.h>

// ---------------- types ----------------
typedef __attribute__((ext_vector_type(16))) __bf16 v16bf;
typedef __attribute__((ext_vector_type(8)))  float  v8f;

#define E_DIM 512
#define H_DIM 8
#define D_DIM 64
#define B_DIM 2
#define S_DIM 4096

// ---------------- CDNA5 async global->LDS path (probe via __has_builtin) ----
#if defined(__has_builtin)
#if __has_builtin(__builtin_amdgcn_global_load_async_to_lds_b128) && \
    __has_builtin(__builtin_amdgcn_s_wait_asynccnt)
#define HAS_ASYNC_LDS 1
#endif
#endif
#ifndef HAS_ASYNC_LDS
#define HAS_ASYNC_LDS 0
#endif

#if HAS_ASYNC_LDS
// Builtin signature (from clang diagnostic): (v4i addrspace(1)*, v4i addrspace(3)*, Ii, Ii)
typedef int v4i_gcc __attribute__((vector_size(4 * sizeof(int))));
typedef __attribute__((address_space(1))) v4i_gcc as1_v4i;
typedef __attribute__((address_space(3))) v4i_gcc as3_v4i;
#endif

// One 16-byte global->LDS transfer. Async (ASYNCcnt) when available.
__device__ inline void ld_g2l_b128(const __bf16* g, __bf16* l) {
#if HAS_ASYNC_LDS
    // Low 32 bits of a generic LDS pointer are the wave-relative LDS byte
    // address (ISA 10.2 aperture rules), which is what AS3 expects.
    __builtin_amdgcn_global_load_async_to_lds_b128(
        (as1_v4i*)(unsigned long long)(size_t)g,
        (as3_v4i*)(unsigned int)(size_t)l,
        0, 0);
#else
    *reinterpret_cast<uint4*>(l) = *reinterpret_cast<const uint4*>(g);
#endif
}

// ---------------- WMMA helper ----------------
__device__ inline v8f wmma_bf16(v16bf a, v16bf b, v8f c) {
    return __builtin_amdgcn_wmma_f32_16x16x32_bf16(
        false, a, false, b, (short)0, c, false, false);
}

// Load a 16x32 bf16 fragment (A layout; B mirrored with lane = column) from a
// row-major [16][K] source (global or LDS).
__device__ inline v16bf load_frag(const __bf16* base, int stride) {
    union { uint4 u[2]; v16bf v; } t;
    const int lane = threadIdx.x & 31;
    const int r    = lane & 15;
    const int hk   = (lane >> 4) << 3;            // 0 or 8
    const __bf16* p = base + (size_t)r * stride + hk;
    t.u[0] = *reinterpret_cast<const uint4*>(p);       // K hk..hk+7
    t.u[1] = *reinterpret_cast<const uint4*>(p + 16);  // K hk+16..hk+23
    return t.v;
}

// half-wave (16-lane) reductions: C-layout rows live across 16 lanes
__device__ inline float half_max16(float v) {
    #pragma unroll
    for (int m = 1; m < 16; m <<= 1) v = fmaxf(v, __shfl_xor(v, m, 32));
    return v;
}
__device__ inline float half_sum16(float v) {
    #pragma unroll
    for (int m = 1; m < 16; m <<= 1) v += __shfl_xor(v, m, 32);
    return v;
}

// ---------------- kernel 1: f32 -> bf16 convert ----------------
__global__ __launch_bounds__(256) void cvt_bf16_kernel(const float* __restrict__ src,
                                                       __bf16* __restrict__ dst, int n) {
    int i = blockIdx.x * blockDim.x + threadIdx.x;
    if (i < n) dst[i] = (__bf16)src[i];
}

// ---------------- kernel 2: fused QKV projection ----------------
__global__ __launch_bounds__(256) void qkv_gemm_kernel(
    const __bf16* __restrict__ xb,
    const __bf16* __restrict__ Wq, const __bf16* __restrict__ Wk, const __bf16* __restrict__ Wv,
    const float* __restrict__ bq, const float* __restrict__ bk, const float* __restrict__ bv,
    __bf16* __restrict__ Qb, __bf16* __restrict__ Kb, __bf16* __restrict__ Vt)
{
    const int wave = threadIdx.x >> 5;
    const int lane = threadIdx.x & 31;
    const int col  = lane & 15;
    const int hi   = (lane >> 4) & 1;

    const int gw   = blockIdx.x * 8 + wave;
    const int mat  = gw / (512 * 32);             // 0=Q 1=K 2=V
    const int rem  = gw % (512 * 32);
    const int m0   = (rem >> 5) * 16;
    const int n0   = (rem & 31) * 16;

    const __bf16* W    = (mat == 0) ? Wq : (mat == 1) ? Wk : Wv;
    const float*  bias = (mat == 0) ? bq : (mat == 1) ? bk : bv;

    v8f acc = {};
    #pragma unroll 4
    for (int k = 0; k < E_DIM; k += 32) {
        v16bf a  = load_frag(xb + (size_t)m0 * E_DIM + k, E_DIM);
        v16bf bf = load_frag(W  + (size_t)n0 * E_DIM + k, E_DIM);
        acc = wmma_bf16(a, bf, acc);
    }

    #pragma unroll
    for (int i = 0; i < 8; ++i) {
        const int m = m0 + i + hi * 8;
        const int n = n0 + col;
        const float v = acc[i] + bias[n];
        const int b = m >> 12, s = m & (S_DIM - 1);
        const int h = n >> 6,  d = n & (D_DIM - 1);
        const size_t bh = (size_t)(b * H_DIM + h);
        if (mat == 2) {
            Vt[(bh * D_DIM + d) * S_DIM + s] = (__bf16)v;
        } else {
            const size_t idx = (bh * S_DIM + s) * D_DIM + d;
            if (mat == 0) Qb[idx] = (__bf16)v; else Kb[idx] = (__bf16)v;
        }
    }
}

// ---------------- kernel 3: flash attention, batch-fused ----------------
// grid: H * (S/128) blocks; 8 waves/block; each wave owns one 16-row q-tile
// for BOTH batches (bias/mask tiles read once, serving b=0 and b=1).
// K/V chunks staged into LDS once per block (async), double-buffered.
__global__ __launch_bounds__(256) void attn_kernel(
    const __bf16* __restrict__ Qb, const __bf16* __restrict__ Kb, const __bf16* __restrict__ Vt,
    const float* __restrict__ bias, const int* __restrict__ mask, __bf16* __restrict__ Ob)
{
    __shared__ __bf16 sK[2][2 * 32 * D_DIM];      // [slot][b][32 k-rows][64 d]  16 KB
    __shared__ __bf16 sV[2][2 * D_DIM * 32];      // [slot][b][64 d-rows][32 k]  16 KB
    __shared__ __bf16 lds_p[8][16 * 32];          // per-wave P transpose buffer  8 KB

    const int tid  = threadIdx.x;
    const int wave = tid >> 5;
    const int lane = tid & 31;
    const int col  = lane & 15;
    const int hi   = (lane >> 4) & 1;

    const int h    = blockIdx.x >> 5;             // 0..7
    const int qblk = blockIdx.x & 31;
    const int q0   = qblk * 128 + wave * 16;

    const __bf16* Qh[2]; const __bf16* Kh[2]; const __bf16* Vh[2];
    #pragma unroll
    for (int bb = 0; bb < 2; ++bb) {
        const size_t hoff = (size_t)(bb * H_DIM + h) * S_DIM * D_DIM;
        Qh[bb] = Qb + hoff; Kh[bb] = Kb + hoff; Vh[bb] = Vt + hoff;
    }

    v16bf qa0[2], qa1[2];
    #pragma unroll
    for (int bb = 0; bb < 2; ++bb) {
        qa0[bb] = load_frag(Qh[bb] + (size_t)q0 * D_DIM,      D_DIM);
        qa1[bb] = load_frag(Qh[bb] + (size_t)q0 * D_DIM + 32, D_DIM);
    }

    v8f acc[2][4];
    float rm[2][8], rl[2][8];
    #pragma unroll
    for (int bb = 0; bb < 2; ++bb) {
        #pragma unroll
        for (int t = 0; t < 4; ++t) acc[bb][t] = v8f{};
        #pragma unroll
        for (int i = 0; i < 8; ++i) { rm[bb][i] = -3.0e38f; rl[bb][i] = 0.0f; }
    }

    const float* bias_p = bias + ((size_t)h * S_DIM + (q0 + hi * 8)) * S_DIM;
    const int*   mask_p = mask + (size_t)(q0 + hi * 8) * S_DIM;

    // ---- stage one 32-wide K/V chunk (both batches) into LDS slot ----
    auto stage = [&](int kc, int slot) {
        #pragma unroll
        for (int j = 0; j < 2; ++j) {
            const int idx  = tid + j * 256;       // 0..511
            const int bsel = idx >> 8;
            const int rem  = idx & 255;
            {   // K tile: [32 rows][64 d], 8 b128 per row
                const int row = rem >> 3, c = (rem & 7) << 3;
                ld_g2l_b128(Kh[bsel] + (size_t)(kc + row) * D_DIM + c,
                            &sK[slot][(bsel * 32 + row) * D_DIM + c]);
            }
            {   // V tile: [64 d-rows][32 k], 4 b128 per row
                const int row = rem >> 2, c = (rem & 3) << 3;
                ld_g2l_b128(Vh[bsel] + (size_t)row * S_DIM + kc + c,
                            &sV[slot][(bsel * D_DIM + row) * 32 + c]);
            }
        }
    };

    stage(0, 0);
    int cur = 0;

    for (int kc = 0; kc < S_DIM; kc += 32) {
        const bool more = (kc + 32) < S_DIM;
        if (more) stage(kc + 32, cur ^ 1);        // prefetch next chunk
#if HAS_ASYNC_LDS
        if (more) __builtin_amdgcn_s_wait_asynccnt(4);   // keep next chunk in flight
        else      __builtin_amdgcn_s_wait_asynccnt(0);
#endif
        __syncthreads();

        // bias/mask tile: loaded once, used for both batches
        float bv0[8], bv1[8]; int mv0[8], mv1[8];
        #pragma unroll
        for (int i = 0; i < 8; ++i) {
            bv0[i] = bias_p[(size_t)i * S_DIM + kc + col];
            bv1[i] = bias_p[(size_t)i * S_DIM + kc + 16 + col];
            mv0[i] = mask_p[(size_t)i * S_DIM + kc + col];
            mv1[i] = mask_p[(size_t)i * S_DIM + kc + 16 + col];
        }

        #pragma unroll
        for (int bb = 0; bb < 2; ++bb) {
            const __bf16* kb = &sK[cur][bb * 32 * D_DIM];   // [32][64]
            const __bf16* vb = &sV[cur][bb * D_DIM * 32];   // [64][32]

            v8f s0 = {}, s1 = {};
            s0 = wmma_bf16(qa0[bb], load_frag(kb,                 D_DIM), s0);
            s0 = wmma_bf16(qa1[bb], load_frag(kb + 32,            D_DIM), s0);
            s1 = wmma_bf16(qa0[bb], load_frag(kb + 16 * D_DIM,      D_DIM), s1);
            s1 = wmma_bf16(qa1[bb], load_frag(kb + 16 * D_DIM + 32, D_DIM), s1);

            float f[8];
            #pragma unroll
            for (int i = 0; i < 8; ++i) {
                float v0 = s0[i] + bv0[i]; v0 = mv0[i] ? v0 : -1.0e20f; v0 *= 0.125f;
                float v1 = s1[i] + bv1[i]; v1 = mv1[i] ? v1 : -1.0e20f; v1 *= 0.125f;

                const float mx   = half_max16(fmaxf(v0, v1));
                const float mnew = fmaxf(rm[bb][i], mx);
                const float p0 = __expf(v0 - mnew);
                const float p1 = __expf(v1 - mnew);
                f[i]      = __expf(rm[bb][i] - mnew);
                rm[bb][i] = mnew;
                rl[bb][i] = rl[bb][i] * f[i] + half_sum16(p0 + p1);

                const int row = i + hi * 8;
                lds_p[wave][row * 32 + col]      = (__bf16)p0;
                lds_p[wave][row * 32 + col + 16] = (__bf16)p1;
            }
            #pragma unroll
            for (int i = 0; i < 8; ++i) {
                acc[bb][0][i] *= f[i]; acc[bb][1][i] *= f[i];
                acc[bb][2][i] *= f[i]; acc[bb][3][i] *= f[i];
            }

            __builtin_amdgcn_wave_barrier();      // wave-private LDS round trip
            const v16bf pa = load_frag(&lds_p[wave][0], 32);
            __builtin_amdgcn_wave_barrier();

            acc[bb][0] = wmma_bf16(pa, load_frag(vb + 0 * 16 * 32, 32), acc[bb][0]);
            acc[bb][1] = wmma_bf16(pa, load_frag(vb + 1 * 16 * 32, 32), acc[bb][1]);
            acc[bb][2] = wmma_bf16(pa, load_frag(vb + 2 * 16 * 32, 32), acc[bb][2]);
            acc[bb][3] = wmma_bf16(pa, load_frag(vb + 3 * 16 * 32, 32), acc[bb][3]);
        }

        __syncthreads();                          // all waves done with slot `cur`
        cur ^= 1;
    }

    // ---- normalize + store bf16 attn output [B][S][E] ----
    #pragma unroll
    for (int bb = 0; bb < 2; ++bb) {
        #pragma unroll
        for (int i = 0; i < 8; ++i) {
            const float inv = 1.0f / rl[bb][i];
            const int row = q0 + i + hi * 8;
            const size_t o = ((size_t)bb * S_DIM + row) * E_DIM + h * D_DIM + col;
            Ob[o + 0]  = (__bf16)(acc[bb][0][i] * inv);
            Ob[o + 16] = (__bf16)(acc[bb][1][i] * inv);
            Ob[o + 32] = (__bf16)(acc[bb][2][i] * inv);
            Ob[o + 48] = (__bf16)(acc[bb][3][i] * inv);
        }
    }
}

// ---------------- kernel 4: output projection (f32 out) ----------------
__global__ __launch_bounds__(256) void oproj_gemm_kernel(
    const __bf16* __restrict__ Ob, const __bf16* __restrict__ Wo,
    const float* __restrict__ bo, float* __restrict__ out)
{
    const int wave = threadIdx.x >> 5;
    const int lane = threadIdx.x & 31;
    const int col  = lane & 15;
    const int hi   = (lane >> 4) & 1;

    const int gw = blockIdx.x * 8 + wave;
    const int m0 = (gw >> 5) * 16;
    const int n0 = (gw & 31) * 16;

    v8f acc = {};
    #pragma unroll 4
    for (int k = 0; k < E_DIM; k += 32) {
        v16bf a  = load_frag(Ob + (size_t)m0 * E_DIM + k, E_DIM);
        v16bf bf = load_frag(Wo + (size_t)n0 * E_DIM + k, E_DIM);
        acc = wmma_bf16(a, bf, acc);
    }
    #pragma unroll
    for (int i = 0; i < 8; ++i) {
        const int m = m0 + i + hi * 8;
        const int n = n0 + col;
        out[(size_t)m * E_DIM + n] = acc[i] + bo[n];
    }
}

// ---------------- host launcher ----------------
extern "C" void kernel_launch(void* const* d_in, const int* in_sizes, int n_in,
                              void* d_out, int out_size, void* d_ws, size_t ws_size,
                              hipStream_t stream) {
    const float* x    = (const float*)d_in[0];
    const float* bias = (const float*)d_in[1];
    const int*   mask = (const int*)d_in[2];
    const float* Wq   = (const float*)d_in[3];
    const float* bq   = (const float*)d_in[4];
    const float* Wk   = (const float*)d_in[5];
    const float* bk   = (const float*)d_in[6];
    const float* Wv   = (const float*)d_in[7];
    const float* bv   = (const float*)d_in[8];
    const float* Wo   = (const float*)d_in[9];
    const float* bo   = (const float*)d_in[10];
    float* out = (float*)d_out;

    constexpr size_t N_X = (size_t)B_DIM * S_DIM * E_DIM;
    constexpr size_t N_W = (size_t)E_DIM * E_DIM;
    constexpr size_t N_H = (size_t)B_DIM * H_DIM * S_DIM * D_DIM;

    char* ws = (char*)d_ws;
    size_t off = 0;
    __bf16* xb  = (__bf16*)(ws + off); off += N_X * 2;
    __bf16* Wqb = (__bf16*)(ws + off); off += N_W * 2;
    __bf16* Wkb = (__bf16*)(ws + off); off += N_W * 2;
    __bf16* Wvb = (__bf16*)(ws + off); off += N_W * 2;
    __bf16* Wob = (__bf16*)(ws + off); off += N_W * 2;
    __bf16* Qb  = (__bf16*)(ws + off); off += N_H * 2;
    __bf16* Kb  = (__bf16*)(ws + off); off += N_H * 2;
    __bf16* Vt  = (__bf16*)(ws + off); off += N_H * 2;
    __bf16* Ob  = (__bf16*)(ws + off); off += N_X * 2;
    (void)ws_size; (void)n_in; (void)in_sizes; (void)out_size;

    cvt_bf16_kernel<<<(int)((N_X + 255) / 256), 256, 0, stream>>>(x,  xb,  (int)N_X);
    cvt_bf16_kernel<<<(int)((N_W + 255) / 256), 256, 0, stream>>>(Wq, Wqb, (int)N_W);
    cvt_bf16_kernel<<<(int)((N_W + 255) / 256), 256, 0, stream>>>(Wk, Wkb, (int)N_W);
    cvt_bf16_kernel<<<(int)((N_W + 255) / 256), 256, 0, stream>>>(Wv, Wvb, (int)N_W);
    cvt_bf16_kernel<<<(int)((N_W + 255) / 256), 256, 0, stream>>>(Wo, Wob, (int)N_W);

    qkv_gemm_kernel<<<(3 * 512 * 32) / 8, 256, 0, stream>>>(
        xb, Wqb, Wkb, Wvb, bq, bk, bv, Qb, Kb, Vt);

    // batch-fused attention: H * (S/128) blocks
    attn_kernel<<<H_DIM * (S_DIM / 128), 256, 0, stream>>>(
        Qb, Kb, Vt, bias, mask, Ob);

    oproj_gemm_kernel<<<(512 * 32) / 8, 256, 0, stream>>>(Ob, Wob, bo, out);
}
